// PackedBinaryConv2d_45432164057650
// MI455X (gfx1250) — 8-bit matrix operand order repaired and checked by reading, compiled, not run on hardware
//
#include <hip/hip_runtime.h>

typedef __attribute__((ext_vector_type(8))) int v8i;

#define C_IN   128
#define H_IN   56
#define W_IN   56
#define N_B    32
#define M_OUT  256
#define K_TOT  (C_IN * 9)
#define HW     (H_IN * W_IN)
#define NPIX   (N_B * HW)

#define BM      128
#define BN      128
#define KB      64
#define THREADS 256

__device__ __forceinline__ int sgn8(float v) { return v >= 0.f ? 1 : -1; }

__device__ __forceinline__ int pack4(int b0, int b1, int b2, int b3) {
    return (b0 & 0xFF) | ((b1 & 0xFF) << 8) | ((b2 & 0xFF) << 16) | ((b3 & 0xFF) << 24);
}

__global__ __launch_bounds__(THREADS)
void quant_x_nhwc(const float* __restrict__ x, unsigned char* __restrict__ xq)
{
    const int p = blockIdx.x * blockDim.x + threadIdx.x;
    const int n  = p / HW;
    const int hw = p - n * HW;
    const float* src = x + (size_t)n * C_IN * HW + hw;
    int4* dst = reinterpret_cast<int4*>(xq + (size_t)p * C_IN);

    for (int cb = 0; cb < C_IN; cb += 16) {
        int pk[4];
#pragma unroll
        for (int q = 0; q < 4; ++q) {
            const int c = cb + q * 4;
            pk[q] = pack4(sgn8(src[(size_t)(c + 0) * HW]),
                          sgn8(src[(size_t)(c + 1) * HW]),
                          sgn8(src[(size_t)(c + 2) * HW]),
                          sgn8(src[(size_t)(c + 3) * HW]));
        }
        dst[cb >> 4] = make_int4(pk[0], pk[1], pk[2], pk[3]);
    }
}

__global__ __launch_bounds__(THREADS)
void quant_w_tapmajor(const float* __restrict__ wgt, unsigned char* __restrict__ wq)
{
    const int idx = blockIdx.x * blockDim.x + threadIdx.x;
    if (idx >= M_OUT * 9) return;
    const int m = idx / 9;
    const int t = idx - m * 9;
    const float* src = wgt + (size_t)m * K_TOT + t;
    int4* dst = reinterpret_cast<int4*>(wq + (size_t)m * K_TOT + t * C_IN);

    for (int cb = 0; cb < C_IN; cb += 16) {
        int pk[4];
#pragma unroll
        for (int q = 0; q < 4; ++q) {
            const int c = cb + q * 4;
            pk[q] = pack4(sgn8(src[(c + 0) * 9]), sgn8(src[(c + 1) * 9]),
                          sgn8(src[(c + 2) * 9]), sgn8(src[(c + 3) * 9]));
        }
        dst[cb >> 4] = make_int4(pk[0], pk[1], pk[2], pk[3]);
    }
}

__global__ __launch_bounds__(THREADS)
void binconv_wmma_iu8(const unsigned char* __restrict__ xq,
                      const unsigned char* __restrict__ wq,
                      float* __restrict__ out)
{
    __shared__ __align__(16) unsigned char sA[BM * KB];
    __shared__ __align__(16) unsigned char sB[BN * KB];

    const int tid  = threadIdx.x;
    const int lane = tid & 31;
    const int wave = tid >> 5;
    const int g    = lane >> 4;
    const int ln16 = lane & 15;

    const int m0 = blockIdx.x * BM;
    const int p0 = blockIdx.y * BN;

    const int waveM = wave & 3;
    const int waveN = wave >> 2;

    const int r  = tid >> 1;
    const int hh = tid & 1;

    const int pix = p0 + r;
    const int n   = pix / HW;
    const int hw  = pix - n * HW;
    const int oh  = hw / W_IN;
    const int ow  = hw - oh * W_IN;

    const unsigned char* aSrcBase = wq + (size_t)(m0 + r) * K_TOT + hh * 32;
    unsigned char* aDst = sA + r * KB + hh * 32;
    unsigned char* bDst = sB + r * KB + hh * 32;

    v8i acc[2][4];
#pragma unroll
    for (int ti = 0; ti < 2; ++ti)
#pragma unroll
        for (int tj = 0; tj < 4; ++tj)
            acc[ti][tj] = (v8i){0, 0, 0, 0, 0, 0, 0, 0};

    for (int kb = 0; kb < K_TOT; kb += KB) {
        const int t    = kb >> 7;
        const int coff = kb & 127;
        const int kh   = t / 3;
        const int kw   = t - kh * 3;

        {
            const int4* src = reinterpret_cast<const int4*>(aSrcBase + kb);
            if (kb + KB < K_TOT)
                __builtin_prefetch(aSrcBase + kb + KB, 0, 1);
            int4 a0 = src[0];
            int4 a1 = src[1];
            int4* dst = reinterpret_cast<int4*>(aDst);
            dst[0] = a0;
            dst[1] = a1;
        }

        {
            const int ih = oh + kh - 1;
            const int iw = ow + kw - 1;
            int4 b0 = make_int4(0, 0, 0, 0), b1 = make_int4(0, 0, 0, 0);
            if ((unsigned)ih < (unsigned)H_IN && (unsigned)iw < (unsigned)W_IN) {
                const int4* src = reinterpret_cast<const int4*>(
                    xq + ((size_t)(n * H_IN + ih) * W_IN + iw) * C_IN + coff + hh * 32);
                b0 = src[0];
                b1 = src[1];
            }
            int4* dst = reinterpret_cast<int4*>(bDst);
            dst[0] = b0;
            dst[1] = b1;
        }

        __syncthreads();

        v8i afrag[2];
#pragma unroll
        for (int ti = 0; ti < 2; ++ti) {
            const unsigned char* row = sA + (waveM * 32 + ti * 16 + ln16) * KB + g * 8;
            int2 a0 = *reinterpret_cast<const int2*>(row + 0);
            int2 a1 = *reinterpret_cast<const int2*>(row + 16);
            int2 a2 = *reinterpret_cast<const int2*>(row + 32);
            int2 a3 = *reinterpret_cast<const int2*>(row + 48);
            v8i a;
            a[0] = a0.x; a[1] = a0.y; a[2] = a1.x; a[3] = a1.y;
            a[4] = a2.x; a[5] = a2.y; a[6] = a3.x; a[7] = a3.y;
            afrag[ti] = a;
        }
        v8i bfrag[4];
#pragma unroll
        for (int tj = 0; tj < 4; ++tj) {
            const unsigned char* row = sB + (waveN * 64 + tj * 16 + ln16) * KB;
            int2 b0 = *reinterpret_cast<const int2*>(row + g * 8 + 0);
            int2 b1 = *reinterpret_cast<const int2*>(row + g * 8 + 16);
            int2 b2 = *reinterpret_cast<const int2*>(row + g * 8 + 32);
            int2 b3 = *reinterpret_cast<const int2*>(row + g * 8 + 48);
            v8i bf;
            bf[0] = b0.x; bf[1] = b0.y; bf[2] = b1.x; bf[3] = b1.y;
            bf[4] = b2.x; bf[5] = b2.y; bf[6] = b3.x; bf[7] = b3.y;
            bfrag[tj] = bf;
        }

#pragma unroll
        for (int ti = 0; ti < 2; ++ti)
#pragma unroll
            for (int tj = 0; tj < 4; ++tj)
                acc[ti][tj] = __builtin_amdgcn_wmma_i32_16x16x64_iu8(
                    true, afrag[ti], true, bfrag[tj],
                    acc[ti][tj], false, false);

        __syncthreads();
    }

#pragma unroll
    for (int tj = 0; tj < 4; ++tj) {
        const int np  = p0 + waveN * 64 + tj * 16 + ln16;
        const int on  = np / HW;
        const int ohw = np - on * HW;
#pragma unroll
        for (int ti = 0; ti < 2; ++ti) {
            const int mbase = m0 + waveM * 32 + ti * 16 + g * 8;
#pragma unroll
            for (int v = 0; v < 8; ++v) {
                out[((size_t)on * M_OUT + (mbase + v)) * HW + ohw] =
                    (float)acc[ti][tj][v];
            }
        }
    }
}

extern "C" void kernel_launch(void* const* d_in, const int* in_sizes, int n_in,
                              void* d_out, int out_size, void* d_ws, size_t ws_size,
                              hipStream_t stream) {
    (void)in_sizes; (void)n_in; (void)out_size; (void)ws_size;
    const float* x   = (const float*)d_in[0];
    const float* wgt = (const float*)d_in[1];
    float* out       = (float*)d_out;

    unsigned char* xq = (unsigned char*)d_ws;
    unsigned char* wq = xq + (size_t)NPIX * C_IN;

    quant_x_nhwc<<<NPIX / THREADS, THREADS, 0, stream>>>(x, xq);
    quant_w_tapmajor<<<(M_OUT * 9 + THREADS - 1) / THREADS, THREADS, 0, stream>>>(wgt, wq);
    dim3 grid(M_OUT / BM, NPIX / BN);
    binconv_wmma_iu8<<<grid, THREADS, 0, stream>>>(xq, wq, out);
}
